// CrossAttention_21225728377535
// MI455X (gfx1250) — matrix operand order repaired and checked by reading, not compiled, not run on hardware
//
#include <hip/hip_runtime.h>

// ---------------------------------------------------------------------------
// CrossAttention forward for MI455X (gfx1250, wave32, WMMA bf16)
//   q  = (X Wq) * hd^-0.5          -> qcache  (B,nh,S,hd)  bf16
//   kv = X_kv Wkv                  -> kcache  (B,nh,T,hd)  bf16
//                                     vtcache (B,nh,hd,T)  bf16 (transposed)
//   y  = softmax(q k^T) v          -> ycache  (B*S, H)     bf16
//   out = y Wc                     -> d_out   fp32
// Matmuls on v_wmma_f32_16x16x32_bf16; global->LDS staging uses the CDNA5
// async-tensor path (global_load_async_to_lds_b128 + s_wait_asynccnt) with
// double-buffered LDS tiles so fetch overlaps WMMA work.
// ---------------------------------------------------------------------------

typedef unsigned short u16;
typedef unsigned int   u32;
typedef __attribute__((ext_vector_type(16))) __bf16 v16bf;
typedef __attribute__((ext_vector_type(8)))  float  v8f;
typedef __attribute__((ext_vector_type(8)))  u32    v8u;

static constexpr int Sdim = 2048;
static constexpr int Tdim = 2048;
static constexpr int Hdim = 1024;
static constexpr int NH   = 16;
static constexpr int HD   = 64;

__device__ __forceinline__ u16 f2bf(float f) {
  u32 u = __float_as_uint(f);
  return (u16)((u + 0x7FFFu + ((u >> 16) & 1u)) >> 16);   // round-to-nearest-even
}

// Build a 16-element bf16 fragment (8 VGPRs) from two 16-byte LDS chunks.
__device__ __forceinline__ v16bf make_frag(const u16* p0, const u16* p1) {
  uint4 a = *reinterpret_cast<const uint4*>(p0);
  uint4 b = *reinterpret_cast<const uint4*>(p1);
  v8u t;
  t[0] = a.x; t[1] = a.y; t[2] = a.z; t[3] = a.w;
  t[4] = b.x; t[5] = b.y; t[6] = b.z; t[7] = b.w;
  return __builtin_bit_cast(v16bf, t);
}

__device__ __forceinline__ v8f wmma_bf(v16bf a, v16bf b, v8f c) {
  return __builtin_amdgcn_wmma_f32_16x16x32_bf16(false, a, false, b, (short)0, c,
                                                 false, false);
}

// LDS byte offset of a generic pointer to __shared__ (low 32 bits of the
// shared-aperture flat address).
__device__ __forceinline__ u32 lds_off(const void* p) {
  return (u32)(uintptr_t)p;
}

// Async global->LDS copies (ASYNCcnt-tracked, no data VGPRs). INST_OFFSET is
// applied to both the memory and the LDS address, so one lds VGPR serves all.
__device__ __forceinline__ void async_copy_32B(u32 lds, const u16* g) {
  asm volatile(
      "global_load_async_to_lds_b128 %0, %1, off\n\t"
      "global_load_async_to_lds_b128 %0, %1, off offset:16"
      :: "v"(lds), "v"(g) : "memory");
}
__device__ __forceinline__ void async_copy_64B(u32 lds, const u16* g) {
  asm volatile(
      "global_load_async_to_lds_b128 %0, %1, off\n\t"
      "global_load_async_to_lds_b128 %0, %1, off offset:16\n\t"
      "global_load_async_to_lds_b128 %0, %1, off offset:32\n\t"
      "global_load_async_to_lds_b128 %0, %1, off offset:48"
      :: "v"(lds), "v"(g) : "memory");
}

// ------------------------------- converters --------------------------------

__global__ __launch_bounds__(256) void f32_to_bf16_kernel(const float* __restrict__ in,
                                                          u16* __restrict__ out, int n) {
  int i = blockIdx.x * 256 + threadIdx.x;
  if (i < n) out[i] = f2bf(in[i]);
}

// in: (K x N) fp32 row-major  ->  out: (N x K) bf16 row-major (i.e. W^T)
__global__ __launch_bounds__(256) void wT_bf16_kernel(const float* __restrict__ in,
                                                      u16* __restrict__ out,
                                                      int K, int N) {
  int k = blockIdx.x * 256 + threadIdx.x;
  int n = blockIdx.y;
  if (k < K) out[(size_t)n * K + k] = f2bf(in[(size_t)k * N + n]);
}

// --------------------------------- GEMM ------------------------------------
// C(MxN) = A(MxK,bf16 rm) * B, with B supplied transposed as BT(NxK,bf16 rm).
// Block tile 128x128, K-step 64, double-buffered async LDS staging.
// 8 waves, each wave owns a 64x32 sub-tile (16 WMMAs per K-step).
// MODE 0: q-projection  -> D0 = qcache (scaled by 1/8)
// MODE 1: kv-projection -> D0 = kcache, D1 = vtcache (transposed V)
// MODE 2: out-projection-> Co = fp32 output
template <int MODE>
__global__ __launch_bounds__(256) void gemm_bf16_kernel(const u16* __restrict__ A,
                                                        const u16* __restrict__ BT,
                                                        float* __restrict__ Co,
                                                        u16* __restrict__ D0,
                                                        u16* __restrict__ D1,
                                                        int M, int N, int K) {
  __shared__ alignas(16) u16 As[2][128 * 64];   // 16 KB x2
  __shared__ alignas(16) u16 Bs[2][128 * 64];   // 16 KB x2

  const int tid  = threadIdx.x;
  const int lane = tid & 31;
  const int w    = tid >> 5;
  const int nl   = lane & 15;     // column / row-within-16
  const int kh   = lane >> 4;     // K half-select
  const int wm   = w >> 2;        // wave row (0..1) -> 64 rows
  const int wn   = w & 3;         // wave col (0..3) -> 32 cols
  const int m0   = blockIdx.y * 128;
  const int n0   = blockIdx.x * 128;

  v8f acc[4][2] = {};

  // staging: each thread owns 64B of A and 64B of B per K-step
  const int srow = tid >> 1, sh = tid & 1;
  const u16* gA = A  + ((size_t)(m0 + srow)) * K + sh * 32;
  const u16* gB = BT + ((size_t)(n0 + srow)) * K + sh * 32;
  const u32 lA[2] = { lds_off(&As[0][srow * 64 + sh * 32]),
                      lds_off(&As[1][srow * 64 + sh * 32]) };
  const u32 lB[2] = { lds_off(&Bs[0][srow * 64 + sh * 32]),
                      lds_off(&Bs[1][srow * 64 + sh * 32]) };

  const int ksteps = K >> 6;
  // prologue: fetch K-step 0 into buffer 0
  async_copy_64B(lA[0], gA);
  async_copy_64B(lB[0], gB);

  for (int kb = 0; kb < ksteps; ++kb) {
    const int cur = kb & 1;
    if (kb + 1 < ksteps) {        // prefetch next K-step into the other buffer
      async_copy_64B(lA[cur ^ 1], gA + (size_t)(kb + 1) * 64);
      async_copy_64B(lB[cur ^ 1], gB + (size_t)(kb + 1) * 64);
      asm volatile("s_wait_asynccnt 8" ::: "memory");   // current buffer done
    } else {
      asm volatile("s_wait_asynccnt 0" ::: "memory");
    }
    __syncthreads();

#pragma unroll
    for (int kbh = 0; kbh < 2; ++kbh) {                 // two 32-deep slices
      v16bf af[4], bfr[2];
#pragma unroll
      for (int mi = 0; mi < 4; ++mi) {                  // A frag: 16x32 rm tile
        const u16* ar = &As[cur][(wm * 64 + mi * 16 + nl) * 64 + kbh * 32 + kh * 8];
        af[mi] = make_frag(ar, ar + 16);
      }
#pragma unroll
      for (int ni = 0; ni < 2; ++ni) {                  // B frag: lane=col, K table order
        const u16* br = &Bs[cur][(wn * 32 + ni * 16 + nl) * 64 + kbh * 32 + kh * 8];
        bfr[ni] = make_frag(br, br + 16);
      }
#pragma unroll
      for (int mi = 0; mi < 4; ++mi)
#pragma unroll
        for (int ni = 0; ni < 2; ++ni)
          acc[mi][ni] = wmma_bf(af[mi], bfr[ni], acc[mi][ni]);
    }
    __syncthreads();              // all reads of buffer `cur` complete
  }

  // Epilogue. C layout: lanes 0-15 -> rows j, lanes 16-31 -> rows j+8; col = nl.
#pragma unroll
  for (int mi = 0; mi < 4; ++mi) {
#pragma unroll
    for (int ni = 0; ni < 2; ++ni) {
#pragma unroll
      for (int j = 0; j < 8; ++j) {
        const int row = m0 + wm * 64 + mi * 16 + j + kh * 8;
        const int col = n0 + wn * 32 + ni * 16 + nl;
        const float v = acc[mi][ni][j];
        if constexpr (MODE == 0) {                 // q: (B,nh,S,hd), scaled
          const int b = row >> 11, s = row & 2047;
          const int h = col >> 6,  d = col & 63;
          D0[(((size_t)(b * NH + h)) * Sdim + s) * HD + d] = f2bf(v * 0.125f);
        } else if constexpr (MODE == 1) {          // k / v caches
          const int b = row >> 11, t = row & 2047;
          if (col < Hdim) {
            const int h = col >> 6, d = col & 63;
            D0[(((size_t)(b * NH + h)) * Tdim + t) * HD + d] = f2bf(v);
          } else {
            const int c2 = col - Hdim;
            const int h = c2 >> 6, d = c2 & 63;
            D1[(((size_t)(b * NH + h)) * HD + d) * Tdim + t] = f2bf(v);
          }
        } else {                                   // fp32 output
          Co[(size_t)row * Hdim + col] = v;
        }
      }
    }
  }
}

// ------------------------- flash attention kernel --------------------------
// One block = (b,h, 128-row s-tile); 8 waves, wave w owns 16 s-rows.
// T is consumed in 64-wide chunks with online softmax; K/V tiles are staged
// with double-buffered async copies; all matmuls in WMMA.
__global__ __launch_bounds__(256) void attn_kernel(const u16* __restrict__ qc,
                                                   const u16* __restrict__ kc,
                                                   const u16* __restrict__ vt,
                                                   u16* __restrict__ yc) {
  __shared__ alignas(16) u16 Ks[2][64 * 64];       // [t][hd]   8 KB x2
  __shared__ alignas(16) u16 Vs[2][64 * 64];       // [hd][t]   8 KB x2
  __shared__ alignas(16) u16 Ps[8 * 16 * 64];      // per-wave P tile [s][t]

  const int tid  = threadIdx.x;
  const int w    = tid >> 5;
  const int lane = tid & 31;
  const int nl   = lane & 15;
  const int kh   = lane >> 4;
  const int bh   = blockIdx.y;                     // b*16 + h
  const int s0   = blockIdx.x * 128;

  // Per-wave q fragments (16 x 64 bf16), kept in registers for the whole loop.
  const u16* qrow = qc + (((size_t)bh * Sdim) + s0 + w * 16 + nl) * HD;
  v16bf qf[2];
  qf[0] = make_frag(qrow +      kh * 8, qrow + 16 +      kh * 8);
  qf[1] = make_frag(qrow + 32 + kh * 8, qrow + 32 + 16 + kh * 8);

  v8f yacc[4] = {};
  float mrun[8], lrun[8];
#pragma unroll
  for (int j = 0; j < 8; ++j) { mrun[j] = -3.0e38f; lrun[j] = 0.f; }

  u16* pb = Ps + w * 1024;

  // staging: 32B of K-tile and 32B of V^T-tile per thread per T-step
  const int srow = tid >> 2, q4 = tid & 3;
  const u16* gK = kc + (((size_t)bh * Tdim) + srow) * HD + q4 * 16;  // + t0*64
  const u16* gV = vt + (((size_t)bh * HD) + srow) * Tdim + q4 * 16;  // + t0
  const u32 lK[2] = { lds_off(&Ks[0][srow * 64 + q4 * 16]),
                      lds_off(&Ks[1][srow * 64 + q4 * 16]) };
  const u32 lV[2] = { lds_off(&Vs[0][srow * 64 + q4 * 16]),
                      lds_off(&Vs[1][srow * 64 + q4 * 16]) };

  async_copy_32B(lK[0], gK);
  async_copy_32B(lV[0], gV);

  for (int t0 = 0; t0 < Tdim; t0 += 64) {
    const int cur = (t0 >> 6) & 1;
    if (t0 + 64 < Tdim) {
      async_copy_32B(lK[cur ^ 1], gK + (size_t)(t0 + 64) * 64);
      async_copy_32B(lV[cur ^ 1], gV + (size_t)(t0 + 64));
      asm volatile("s_wait_asynccnt 4" ::: "memory");
    } else {
      asm volatile("s_wait_asynccnt 0" ::: "memory");
    }
    __syncthreads();

    // scores S = q K^T : 16(s) x 64(t), K-dim = hd = 64
    v8f sacc[4] = {};
#pragma unroll
    for (int kb = 0; kb < 2; ++kb)
#pragma unroll
      for (int ni = 0; ni < 4; ++ni) {
        const u16* br = &Ks[cur][(ni * 16 + nl) * 64 + kb * 32 + kh * 8];
        sacc[ni] = wmma_bf(qf[kb], make_frag(br, br + 16), sacc[ni]);
      }

    // online softmax (row reductions across 16-lane halves)
#pragma unroll
    for (int j = 0; j < 8; ++j) {
      float tmax = fmaxf(fmaxf(sacc[0][j], sacc[1][j]),
                         fmaxf(sacc[2][j], sacc[3][j]));
#pragma unroll
      for (int off = 1; off < 16; off <<= 1)
        tmax = fmaxf(tmax, __shfl_xor(tmax, off));
      const float mnew  = fmaxf(mrun[j], tmax);
      const float alpha = __expf(mrun[j] - mnew);
      float rsum = 0.f;
#pragma unroll
      for (int ni = 0; ni < 4; ++ni) {
        const float p = __expf(sacc[ni][j] - mnew);
        pb[(j + kh * 8) * 64 + ni * 16 + nl] = f2bf(p);   // reshape C->A via LDS
        rsum += p;
      }
#pragma unroll
      for (int off = 1; off < 16; off <<= 1)
        rsum += __shfl_xor(rsum, off);
      lrun[j] = lrun[j] * alpha + rsum;
      mrun[j] = mnew;
#pragma unroll
      for (int ni = 0; ni < 4; ++ni) yacc[ni][j] *= alpha;
    }

    // y += P V : A = P (16 x 64 over t), B^T = Vs rows [hd][t]
#pragma unroll
    for (int kb = 0; kb < 2; ++kb) {
      const u16* pr = pb + nl * 64 + kb * 32 + kh * 8;
      const v16bf pf = make_frag(pr, pr + 16);
#pragma unroll
      for (int ni = 0; ni < 4; ++ni) {
        const u16* vr = &Vs[cur][(ni * 16 + nl) * 64 + kb * 32 + kh * 8];
        yacc[ni] = wmma_bf(pf, make_frag(vr, vr + 16), yacc[ni]);
      }
    }
    __syncthreads();              // done reading buffer `cur`
  }

  // normalize and store y into (B*S, H) bf16 (H index = h*64 + d)
  const int b = bh >> 4, h = bh & 15;
#pragma unroll
  for (int j = 0; j < 8; ++j) {
    const float inv = 1.0f / lrun[j];
    const int srw   = s0 + w * 16 + j + kh * 8;
#pragma unroll
    for (int ni = 0; ni < 4; ++ni)
      yc[(((size_t)b * Sdim) + srw) * Hdim + h * HD + ni * 16 + nl] =
          f2bf(yacc[ni][j] * inv);
  }
}

// ------------------------------- launcher ----------------------------------

extern "C" void kernel_launch(void* const* d_in, const int* in_sizes, int n_in,
                              void* d_out, int out_size, void* d_ws, size_t ws_size,
                              hipStream_t stream) {
  (void)in_sizes; (void)n_in; (void)out_size; (void)ws_size;

  const float* query = (const float*)d_in[0];   // (2,2048,1024)
  const float* keyv  = (const float*)d_in[1];   // (2,2048,1024)
  const float* Wq    = (const float*)d_in[2];   // (1024,1024)
  const float* Wkv   = (const float*)d_in[3];   // (1024,2048)
  const float* Wc    = (const float*)d_in[4];   // (1024,1024)
  float* out = (float*)d_out;

  // workspace carve-up (bf16 elements), ~56 MB total
  u16* p    = (u16*)d_ws;
  u16* qx   = p; p += (size_t)4096 * 1024;   // query bf16
  u16* kvx  = p; p += (size_t)4096 * 1024;   // key_value bf16
  u16* WqT  = p; p += (size_t)1024 * 1024;
  u16* WkvT = p; p += (size_t)2048 * 1024;
  u16* WcT  = p; p += (size_t)1024 * 1024;
  u16* qcac = p; p += (size_t)4096 * 1024;   // (B,nh,S,hd)
  u16* kcac = p; p += (size_t)4096 * 1024;   // (B,nh,T,hd)
  u16* vtca = p; p += (size_t)4096 * 1024;   // (B,nh,hd,T)
  u16* ycac = p; p += (size_t)4096 * 1024;   // (B*S,H)

  const int nact = 4096 * 1024;
  f32_to_bf16_kernel<<<nact / 256, 256, 0, stream>>>(query, qx, nact);
  f32_to_bf16_kernel<<<nact / 256, 256, 0, stream>>>(keyv, kvx, nact);
  wT_bf16_kernel<<<dim3(4, 1024), 256, 0, stream>>>(Wq,  WqT,  1024, 1024);
  wT_bf16_kernel<<<dim3(4, 2048), 256, 0, stream>>>(Wkv, WkvT, 1024, 2048);
  wT_bf16_kernel<<<dim3(4, 1024), 256, 0, stream>>>(Wc,  WcT,  1024, 1024);

  // q = X Wq (scaled), k/v = X Wkv
  gemm_bf16_kernel<0><<<dim3(8, 32),  256, 0, stream>>>(qx,  WqT,  nullptr, qcac, nullptr, 4096, 1024, 1024);
  gemm_bf16_kernel<1><<<dim3(16, 32), 256, 0, stream>>>(kvx, WkvT, nullptr, kcac, vtca,   4096, 2048, 1024);

  // flash attention: grid = (S/128, B*nh)
  attn_kernel<<<dim3(16, 32), 256, 0, stream>>>(qcac, kcac, vtca, ycac);

  // out = y Wc (fp32)
  gemm_bf16_kernel<2><<<dim3(8, 32), 256, 0, stream>>>(ycac, WcT, out, nullptr, nullptr, 4096, 1024, 1024);
}
